// MergedDeepseekMoE_69544110457105
// MI455X (gfx1250) — compile-verified, matrix operand order checked by reading, not run on hardware
//
#include <hip/hip_runtime.h>
#include <hip/hip_bf16.h>

typedef __attribute__((ext_vector_type(16))) _Float16 v16h;
typedef __attribute__((ext_vector_type(8)))  _Float16 v8h;
typedef __attribute__((ext_vector_type(8)))  float    v8f;
typedef __attribute__((ext_vector_type(2)))  int      v2i;

#define HD   2048        // hidden
#define ID   1408        // moe intermediate
#define ISD  2816        // shared intermediate
#define NE   8           // routed experts
#define KT   14080       // NE*ID + ISD
#define NT   2048        // tokens
#define BM   128
#define BN   64
#define BK   32
#define LDA  40          // 32 halves + 8 pad -> conflict-free b128 frag loads

#if __has_builtin(__builtin_amdgcn_global_load_async_to_lds_b64) && \
    __has_builtin(__builtin_amdgcn_s_wait_asynccnt)
#define HAVE_ASYNC 1
#else
#define HAVE_ASYNC 0
#endif

// pack two fp32 -> two f16 (single v_cvt_pk_rtz_f16_f32)
__device__ __forceinline__ unsigned int pk2h(float a, float b) {
  return __builtin_bit_cast(unsigned int, __builtin_amdgcn_cvt_pkrtz(a, b));
}

__device__ __forceinline__ unsigned short f2h(float f) {
  _Float16 h = (_Float16)f;                 // v_cvt_f16_f32 (RNE)
  return __builtin_bit_cast(unsigned short, h);
}

__device__ __forceinline__ v16h frag_ld(const unsigned short* p) {
  v8h lo = *(const v8h*)(p);                // K klo .. klo+7
  v8h hi = *(const v8h*)(p + 16);           // K klo+16 .. klo+23
  return __builtin_shufflevector(lo, hi, 0,1,2,3,4,5,6,7,8,9,10,11,12,13,14,15);
}

// ---------------------------------------------------------------- gate/top-2
// writes combine9[N][9]: columns 0..7 = scattered top-2 softmax, column 8 = 1.0
__global__ __launch_bounds__(256) void moe_gate(const float* __restrict__ x,
                                                const float* __restrict__ gate_w,
                                                float* __restrict__ combine9) {
  const int tid  = threadIdx.x;
  const int lane = tid & 31;
  const int n    = blockIdx.x * 8 + (tid >> 5);   // wave per token

  float acc[NE];
#pragma unroll
  for (int e = 0; e < NE; ++e) acc[e] = 0.f;

  for (int h = lane; h < HD; h += 32) {
    const float xv = x[n * HD + h];
#pragma unroll
    for (int e = 0; e < NE; ++e) acc[e] += xv * gate_w[e * HD + h];
  }
#pragma unroll
  for (int e = 0; e < NE; ++e) {
#pragma unroll
    for (int off = 16; off >= 1; off >>= 1) acc[e] += __shfl_xor(acc[e], off, 32);
  }

  float m1 = -1e30f, m2 = -1e30f; int i1 = 0, i2 = 0;
#pragma unroll
  for (int e = 0; e < NE; ++e) {
    if (acc[e] > m1) { m2 = m1; i2 = i1; m1 = acc[e]; i1 = e; }
    else if (acc[e] > m2) { m2 = acc[e]; i2 = e; }
  }
  float den = 0.f;
#pragma unroll
  for (int e = 0; e < NE; ++e) den += __expf(acc[e] - m1);
  const float inv = __builtin_amdgcn_rcpf(den);
  const float w1 = inv;
  const float w2 = __expf(m2 - m1) * inv;
  if (lane < 9) {
    float v = (lane == 8) ? 1.f : (lane == i1) ? w1 : (lane == i2) ? w2 : 0.f;
    combine9[n * 9 + lane] = v;
  }
}

// --------------------------------------------- fused gate/up GEMM + SiLU*up
// grid.z: 0..7 routed experts, 8 = shared expert pair
__global__ __launch_bounds__(256) void moe_gateup(const float* __restrict__ x,
                                                  const float* __restrict__ w_gate,
                                                  const float* __restrict__ w_up,
                                                  const float* __restrict__ sw_gate,
                                                  const float* __restrict__ sw_up,
                                                  const float* __restrict__ combine9,
                                                  unsigned short* __restrict__ hbig) {
  const int z = blockIdx.z;
  const float* wg; const float* wu; int ld, ncols, colofs;
  if (z < NE) { wg = w_gate + (size_t)z * HD * ID; wu = w_up + (size_t)z * HD * ID;
                ld = ID;  ncols = ID;  colofs = z * ID; }
  else        { wg = sw_gate; wu = sw_up;
                ld = ISD; ncols = ISD; colofs = NE * ID; }

  const int n0 = blockIdx.y * BN;
  if (n0 >= ncols) return;                       // uniform per block, pre-barrier
  const int m0 = blockIdx.x * BM;

  __shared__ __align__(16) unsigned short sA [BM * LDA];
  __shared__ __align__(16) unsigned short sBg[BN * LDA];
  __shared__ __align__(16) unsigned short sBu[BN * LDA];

  const int tid  = threadIdx.x;
  const int lane = tid & 31;
  const int wv   = tid >> 5;
  const int fm   = lane & 15;
  const int klo  = (lane >> 4) * 8;

  v8f accg[4] = {{}, {}, {}, {}};
  v8f accu[4] = {{}, {}, {}, {}};

  const int nB  = tid & 63;       // B staging: column within tile
  const int kbB = tid >> 6;       // 0..3

  for (int kt = 0; kt < HD; kt += BK) {
    // stage X tile (fp32 -> f16), 128x32
#pragma unroll
    for (int i = 0; i < 4; ++i) {
      const int q = tid + i * 256;               // 1024 float4 loads
      const int row = q >> 3, c4 = q & 7;
      const float4 v = *(const float4*)&x[(size_t)(m0 + row) * HD + kt + c4 * 4];
      *(uint2*)&sA[row * LDA + c4 * 4] = make_uint2(pk2h(v.x, v.y), pk2h(v.z, v.w));
    }
    // stage W tiles transposed: sB[n][k], 32x64 each
#pragma unroll
    for (int j = 0; j < 4; ++j) {
      const int p  = kbB + 4 * j;                // K pair index 0..15
      const int kg = kt + 2 * p;
      const float g0 = wg[(size_t)kg       * ld + n0 + nB];
      const float g1 = wg[(size_t)(kg + 1) * ld + n0 + nB];
      const float u0 = wu[(size_t)kg       * ld + n0 + nB];
      const float u1 = wu[(size_t)(kg + 1) * ld + n0 + nB];
      *(unsigned int*)&sBg[nB * LDA + 2 * p] = pk2h(g0, g1);
      *(unsigned int*)&sBu[nB * LDA + 2 * p] = pk2h(u0, u1);
    }
    __syncthreads();

    const v16h a = frag_ld(&sA[(wv * 16 + fm) * LDA + klo]);
#pragma unroll
    for (int nt = 0; nt < 4; ++nt) {
      v16h bg = frag_ld(&sBg[(nt * 16 + fm) * LDA + klo]);
      v16h bu = frag_ld(&sBu[(nt * 16 + fm) * LDA + klo]);
      accg[nt] = __builtin_amdgcn_wmma_f32_16x16x32_f16(false, a, false, bg, (short)0, accg[nt], false, false);
      accu[nt] = __builtin_amdgcn_wmma_f32_16x16x32_f16(false, a, false, bu, (short)0, accu[nt], false, false);
    }
    __syncthreads();
  }

  // epilogue: silu(g)*u * combine -> f16 Hbig (branch-free scale via 9th col)
  float scale[8];
#pragma unroll
  for (int r = 0; r < 8; ++r) {
    const int m = m0 + wv * 16 + r + ((lane >> 4) * 8);
    scale[r] = combine9[m * 9 + z];
  }
#pragma unroll
  for (int nt = 0; nt < 4; ++nt) {
    const int nc = n0 + nt * 16 + fm;
#pragma unroll
    for (int r = 0; r < 8; ++r) {
      const int m = m0 + wv * 16 + r + ((lane >> 4) * 8);
      const float g = accg[nt][r], u = accu[nt][r];
      const float s = __builtin_amdgcn_rcpf(1.f + __expf(-g));
      hbig[(size_t)m * KT + colofs + nc] = f2h(g * s * u * scale[r]);
    }
  }
}

// -------------------------------------- down GEMM: Y = Hbig @ [w_down;sw_down]
__global__ __launch_bounds__(256) void moe_down(const unsigned short* __restrict__ hbig,
                                                const float* __restrict__ w_down,
                                                const float* __restrict__ sw_down,
                                                float* __restrict__ y) {
  const int m0 = blockIdx.x * BM;
  const int h0 = blockIdx.y * BN;

  __shared__ __align__(16) unsigned short sA[BM * LDA];
  __shared__ __align__(16) unsigned short sB[BN * LDA];

  const int tid  = threadIdx.x;
  const int lane = tid & 31;
  const int wv   = tid >> 5;
  const int fm   = lane & 15;
  const int klo  = (lane >> 4) * 8;

  v8f acc[4] = {{}, {}, {}, {}};

  const int nB  = tid & 63;
  const int kbB = tid >> 6;

  for (int kt = 0; kt < KT; kt += BK) {
    // stage Hbig tile (already f16), 128x32 — CDNA5 async global->LDS copy
#pragma unroll
    for (int i = 0; i < 4; ++i) {
      const int q = tid + i * 256;
      const int row = q >> 3, c = q & 7;
#if HAVE_ASYNC
      __builtin_amdgcn_global_load_async_to_lds_b64(
          (__attribute__((address_space(1))) v2i*)&hbig[(size_t)(m0 + row) * KT + kt + c * 4],
          (__attribute__((address_space(3))) v2i*)&sA[row * LDA + c * 4],
          0, 0);
#else
      const uint2 v = *(const uint2*)&hbig[(size_t)(m0 + row) * KT + kt + c * 4];
      *(uint2*)&sA[row * LDA + c * 4] = v;
#endif
    }
    // stage down-proj rows transposed: sB[n][k], 32x64
#pragma unroll
    for (int j = 0; j < 4; ++j) {
      const int p  = kbB + 4 * j;
      const int kg = kt + 2 * p;                 // pair (kg, kg+1) same source
      float w0, w1;
      if (kg < NE * ID) {
        w0 = w_down[(size_t)kg       * HD + h0 + nB];
        w1 = w_down[(size_t)(kg + 1) * HD + h0 + nB];
      } else {
        w0 = sw_down[(size_t)(kg - NE * ID)     * HD + h0 + nB];
        w1 = sw_down[(size_t)(kg + 1 - NE * ID) * HD + h0 + nB];
      }
      *(unsigned int*)&sB[nB * LDA + 2 * p] = pk2h(w0, w1);
    }
#if HAVE_ASYNC
    __builtin_amdgcn_s_wait_asynccnt(0);
#endif
    __syncthreads();

    const v16h a = frag_ld(&sA[(wv * 16 + fm) * LDA + klo]);
#pragma unroll
    for (int nt = 0; nt < 4; ++nt) {
      v16h b = frag_ld(&sB[(nt * 16 + fm) * LDA + klo]);
      acc[nt] = __builtin_amdgcn_wmma_f32_16x16x32_f16(false, a, false, b, (short)0, acc[nt], false, false);
    }
    __syncthreads();
  }

#pragma unroll
  for (int nt = 0; nt < 4; ++nt) {
#pragma unroll
    for (int r = 0; r < 8; ++r) {
      const int m = m0 + wv * 16 + r + ((lane >> 4) * 8);
      y[(size_t)m * HD + h0 + nt * 16 + fm] = acc[nt][r];
    }
  }
}

extern "C" void kernel_launch(void* const* d_in, const int* in_sizes, int n_in,
                              void* d_out, int out_size, void* d_ws, size_t ws_size,
                              hipStream_t stream) {
  const float* x       = (const float*)d_in[0];  // [1,2048,2048]
  const float* gate_w  = (const float*)d_in[1];  // [8,2048]
  const float* w_gate  = (const float*)d_in[2];  // [8,2048,1408]
  const float* w_up    = (const float*)d_in[3];  // [8,2048,1408]
  const float* w_down  = (const float*)d_in[4];  // [8,1408,2048]
  const float* sw_gate = (const float*)d_in[5];  // [2048,2816]
  const float* sw_up   = (const float*)d_in[6];  // [2048,2816]
  const float* sw_down = (const float*)d_in[7];  // [2816,2048]
  float* y = (float*)d_out;

  float* combine9 = (float*)d_ws;                                  // 2048*9 f32
  unsigned short* hbig = (unsigned short*)((char*)d_ws + 131072);  // 2048*14080 f16

  moe_gate<<<NT / 8, 256, 0, stream>>>(x, gate_w, combine9);

  dim3 gA(NT / BM, ISD / BN, NE + 1);   // (16, 44, 9); expert blocks >= I/BN early-out
  moe_gateup<<<gA, 256, 0, stream>>>(x, w_gate, w_up, sw_gate, sw_up, combine9, hbig);

  dim3 gB(NT / BM, HD / BN);            // (16, 32)
  moe_down<<<gB, 256, 0, stream>>>(hbig, w_down, sw_down, y);
}
